// EdgePredictionGNN_68951404970526
// MI455X (gfx1250) — matrix operand order repaired and checked by reading, not compiled, not run on hardware
//
#include <hip/hip_runtime.h>
#include <hip/hip_bf16.h>
#include <math.h>

// ---------------- problem constants (from reference) ----------------
#define Bg   4
#define Nn   8192
#define Kn   16
#define BN   (Bg * Nn)      // 32768 nodes
#define EMBC 64
#define HIDC 128

typedef __attribute__((ext_vector_type(16))) _Float16 v16h;
typedef __attribute__((ext_vector_type(8)))  _Float16 v8h;
typedef __attribute__((ext_vector_type(8)))  float    v8f;

// ---------------- WMMA operand loaders (CDNA5 16-bit layouts) ----------------
// A 16x32 f16 tile, row-major source with row stride `ld` halves.
// lane<16 : M=lane,    elems 0..7 = K0..7,  elems 8..15 = K16..23
// lane>=16: M=lane-16, elems 0..7 = K8..15, elems 8..15 = K24..31
__device__ __forceinline__ v16h load_a(const _Float16* tile, int ld, int lane) {
  const int m = lane & 15, sel = lane >> 4;
  const _Float16* p = tile + (size_t)m * ld;
  v8h lo = *(const v8h*)(p + sel * 8);
  v8h hi = *(const v8h*)(p + 16 + sel * 8);
  v16h r;
#pragma unroll
  for (int i = 0; i < 8; ++i) { r[i] = lo[i]; r[i + 8] = hi[i]; }
  return r;
}

// B 32x16 f16 tile. Source is "column rows": row n holds B[k][n] for k=0..,
// i.e. transposed weights wt[n][k] (or feature rows x[n][k]), row stride `ld`.
// lane<16 : N=lane, K=0..7,16..23 ; lane>=16: N=lane-16, K=8..15,24..31
__device__ __forceinline__ v16h load_b(const _Float16* tile, int ld, int lane) {
  const int n = lane & 15, sel = lane >> 4;
  const _Float16* p = tile + (size_t)n * ld + sel * 8;
  v8h lo = *(const v8h*)(p);
  v8h hi = *(const v8h*)(p + 16);
  v16h r;
#pragma unroll
  for (int i = 0; i < 8; ++i) { r[i] = lo[i]; r[i + 8] = hi[i]; }
  return r;
}

__device__ __forceinline__ v8f wmma_f16(v16h a, v16h b, v8f c) {
  return __builtin_amdgcn_wmma_f32_16x16x32_f16(false, a, false, b, (short)0, c,
                                                false, false);
}

// ---------------- streaming top-16 (registers only, fully unrolled) ----------------
__device__ __forceinline__ void topk_insert(float (&bd)[16], int (&bi)[16],
                                            float &worst, float d, int id) {
  if (d < worst) {
    int am = 0; float mv = bd[0];
#pragma unroll
    for (int s = 1; s < 16; ++s) { if (bd[s] > mv) { mv = bd[s]; am = s; } }
#pragma unroll
    for (int s = 0; s < 16; ++s) { if (s == am) { bd[s] = d; bi[s] = id; } }
    mv = bd[0];
#pragma unroll
    for (int s = 1; s < 16; ++s) mv = fmaxf(mv, bd[s]);
    worst = mv;
  }
}

__device__ __forceinline__ void sort16(float (&bd)[16], int (&bi)[16]) {
#pragma unroll
  for (int a = 0; a < 15; ++a) {
#pragma unroll
    for (int b = a + 1; b < 16; ++b) {
      bool sw = (bd[b] < bd[a]) || (bd[b] == bd[a] && bi[b] < bi[a]);
      float td0 = sw ? bd[b] : bd[a], td1 = sw ? bd[a] : bd[b];
      int   ti0 = sw ? bi[b] : bi[a], ti1 = sw ? bi[a] : bi[b];
      bd[a] = td0; bd[b] = td1; bi[a] = ti0; bi[b] = ti1;
    }
  }
}

// ---------------- weight transpose + f16 convert: wt[f][c] = w[c][f] ----------------
__global__ void transpose_f16_kernel(const float* __restrict__ w,
                                     _Float16* __restrict__ wt, int C, int F) {
  int t = blockIdx.x * blockDim.x + threadIdx.x;
  if (t < C * F) {
    int f = t / C, c = t % C;
    wt[t] = (_Float16)w[(size_t)c * F + f];
  }
}

// ---------------- kNN on 3-d inputs (scalar; 3 dims too thin for WMMA) --------------
__global__ void knn3_kernel(const float* __restrict__ x, int* __restrict__ idxOut) {
  __shared__ float xc[256][3];
  const int row = blockIdx.x * 256 + threadIdx.x;      // global query node
  const int gBase = (row / Nn) * Nn;
  const float qx = x[(size_t)row * 3 + 0];
  const float qy = x[(size_t)row * 3 + 1];
  const float qz = x[(size_t)row * 3 + 2];

  float bd[16]; int bi[16];
#pragma unroll
  for (int s = 0; s < 16; ++s) { bd[s] = 3.0e38f; bi[s] = 0; }
  float worst = 3.0e38f;

#pragma unroll 1
  for (int chunk = 0; chunk < Nn; chunk += 256) {
    const int c = gBase + chunk + threadIdx.x;
    xc[threadIdx.x][0] = x[(size_t)c * 3 + 0];
    xc[threadIdx.x][1] = x[(size_t)c * 3 + 1];
    xc[threadIdx.x][2] = x[(size_t)c * 3 + 2];
    __syncthreads();
#pragma unroll 1
    for (int j = 0; j < 256; ++j) {
      const int cg = gBase + chunk + j;
      const float dx = qx - xc[j][0], dy = qy - xc[j][1], dz = qz - xc[j][2];
      const float d2 = dx * dx + dy * dy + dz * dz;
      if (cg != row) topk_insert(bd, bi, worst, d2, cg);
    }
    __syncthreads();
  }
  sort16(bd, bi);
#pragma unroll
  for (int s = 0; s < 16; ++s) idxOut[(size_t)row * Kn + s] = bi[s];
}

// ---------------- kNN on 64-d embeddings: fused WMMA dot + streaming top-16 ---------
// 128 threads = 4 waves per block. One wave owns 32 resident query rows (A tiles in
// registers) and streams 64 candidates (4 WMMA column tiles) per LDS round, so the
// two workgroup barriers are amortized over 16 WMMAs. The subtile loop is kept
// rolled (#pragma unroll 1) so only one B-operand/accumulator set is live at a time
// -> no VGPR spills of the resident A operands.
__global__ void knn64_kernel(const _Float16* __restrict__ xh,
                             const float* __restrict__ sq,
                             int* __restrict__ idxOut) {
  __shared__ __align__(16) float dotT[4][32][64];   // 32 KB
  __shared__ float sqcT[4][64];
  const int lane = threadIdx.x & 31;
  const int wv   = threadIdx.x >> 5;                // 0..3
  const int rowBase = blockIdx.x * 128 + wv * 32;
  const int gBase   = (rowBase / Nn) * Nn;
  const int myRow   = rowBase + lane;
  const float sqq   = sq[myRow];
  const int sel = lane >> 4, n = lane & 15;

  const v16h a00 = load_a(xh + (size_t)rowBase * 64,             64, lane);
  const v16h a01 = load_a(xh + (size_t)rowBase * 64 + 32,        64, lane);
  const v16h a10 = load_a(xh + (size_t)(rowBase + 16) * 64,      64, lane);
  const v16h a11 = load_a(xh + (size_t)(rowBase + 16) * 64 + 32, 64, lane);

  float bd[16]; int bi[16];
#pragma unroll
  for (int s = 0; s < 16; ++s) { bd[s] = 3.0e38f; bi[s] = 0; }
  float worst = 3.0e38f;

#pragma unroll 1
  for (int tile = 0; tile < Nn / 64; ++tile) {      // 128 iterations
    const int candBase = gBase + tile * 64;
#pragma unroll 1
    for (int st = 0; st < 4; ++st) {                // 4 x 16-candidate subtiles
      const _Float16* cb = xh + (size_t)(candBase + st * 16) * 64;
      const v16h b0 = load_b(cb,      64, lane);
      const v16h b1 = load_b(cb + 32, 64, lane);
      v8f c0 = {}; v8f c1 = {};
      c0 = wmma_f16(a00, b0, c0);
      c0 = wmma_f16(a01, b1, c0);
      c1 = wmma_f16(a10, b0, c1);
      c1 = wmma_f16(a11, b1, c1);
#pragma unroll
      for (int r = 0; r < 8; ++r) {
        dotT[wv][r + 8 * sel][st * 16 + n]      = c0[r];
        dotT[wv][16 + r + 8 * sel][st * 16 + n] = c1[r];
      }
    }
    sqcT[wv][lane]      = sq[candBase + lane];
    sqcT[wv][lane + 32] = sq[candBase + lane + 32];
    __syncthreads();
#pragma unroll 4
    for (int j = 0; j < 64; ++j) {
      const int cg = candBase + j;
      const float d2 = sqq + sqcT[wv][j] - 2.0f * dotT[wv][lane][j];
      if (cg != myRow) topk_insert(bd, bi, worst, d2, cg);
    }
    __syncthreads();
  }
  sort16(bd, bi);
#pragma unroll
  for (int s = 0; s < 16; ++s) idxOut[(size_t)myRow * Kn + s] = bi[s];
}

// ---------------- EdgeConv1: 6 -> 64 (scalar) -> WMMA 64x64 -> max over K -----------
__global__ void conv1_kernel(const float* __restrict__ x, const int* __restrict__ idx,
                             const float* __restrict__ w1, const float* __restrict__ b1,
                             const _Float16* __restrict__ w2t, const float* __restrict__ b2,
                             float* __restrict__ embOut, _Float16* __restrict__ embhOut,
                             float* __restrict__ sqOut) {
  __shared__ __align__(16) _Float16 feat[8][16][64];
  __shared__ float w1s[6 * 64], b1s[64], b2s[64];
  for (int t = threadIdx.x; t < 6 * 64; t += 256) w1s[t] = w1[t];
  for (int t = threadIdx.x; t < 64; t += 256) { b1s[t] = b1[t]; b2s[t] = b2[t]; }
  __syncthreads();

  const int lane = threadIdx.x & 31, wv = threadIdx.x >> 5;
  const int e = lane & 15, half = lane >> 4;
  const int waveId = blockIdx.x * 8 + wv;

  v16h wB[4][2];
#pragma unroll
  for (int t = 0; t < 4; ++t) {
    wB[t][0] = load_b(w2t + (size_t)t * 16 * 64,      64, lane);
    wB[t][1] = load_b(w2t + (size_t)t * 16 * 64 + 32, 64, lane);
  }

#pragma unroll 1
  for (int it = 0; it < 16; ++it) {
    const int i = waveId * 16 + it;
    const float xi0 = x[(size_t)i * 3 + 0], xi1 = x[(size_t)i * 3 + 1], xi2 = x[(size_t)i * 3 + 2];
    const int j = idx[(size_t)i * Kn + e];
    const float d0 = x[(size_t)j * 3 + 0] - xi0;
    const float d1 = x[(size_t)j * 3 + 1] - xi1;
    const float d2 = x[(size_t)j * 3 + 2] - xi2;
#pragma unroll
    for (int ff = 0; ff < 32; ++ff) {
      const int f = half * 32 + ff;
      float a = b1s[f] + xi0 * w1s[0 * 64 + f] + xi1 * w1s[1 * 64 + f] +
                xi2 * w1s[2 * 64 + f] + d0 * w1s[3 * 64 + f] +
                d1 * w1s[4 * 64 + f] + d2 * w1s[5 * 64 + f];
      feat[wv][e][f] = (_Float16)fmaxf(a, 0.0f);
    }
    __syncthreads();
    const v16h A0 = load_a(&feat[wv][0][0],      64, lane);
    const v16h A1 = load_a(&feat[wv][0][0] + 32, 64, lane);
    float sqa = 0.0f;
#pragma unroll
    for (int t = 0; t < 4; ++t) {
      v8f acc = {};
      acc = wmma_f16(A0, wB[t][0], acc);
      acc = wmma_f16(A1, wB[t][1], acc);
      float mv = acc[0];
#pragma unroll
      for (int r = 1; r < 8; ++r) mv = fmaxf(mv, acc[r]);
      mv = fmaxf(mv, __shfl_xor(mv, 16));
      const int col = t * 16 + e;
      const float outv = mv + b2s[col];
      if (half == 0) {
        embOut[(size_t)i * 64 + col]  = outv;
        embhOut[(size_t)i * 64 + col] = (_Float16)outv;
      }
      sqa += outv * outv;
    }
#pragma unroll
    for (int m = 8; m >= 1; m >>= 1) sqa += __shfl_xor(sqa, m);
    if (lane == 0) sqOut[i] = sqa;
    __syncthreads();
  }
}

// ---------------- EdgeConv2: WMMA 128x64 -> relu -> WMMA 64x64 -> max over K --------
__global__ void conv2_kernel(const float* __restrict__ xin, const int* __restrict__ idx,
                             const _Float16* __restrict__ w1t, const float* __restrict__ b1,
                             const _Float16* __restrict__ w2t, const float* __restrict__ b2,
                             float* __restrict__ embOut, _Float16* __restrict__ embhOut,
                             float* __restrict__ sqOut) {
  __shared__ __align__(16) _Float16 feat[8][16][128];
  __shared__ __align__(16) _Float16 h1s[8][16][64];
  const int lane = threadIdx.x & 31, wv = threadIdx.x >> 5;
  const int e = lane & 15, half = lane >> 4;
  const int waveId = blockIdx.x * 8 + wv;

  v16h wB2[4][2];
#pragma unroll
  for (int t = 0; t < 4; ++t) {
    wB2[t][0] = load_b(w2t + (size_t)t * 16 * 64,      64, lane);
    wB2[t][1] = load_b(w2t + (size_t)t * 16 * 64 + 32, 64, lane);
  }

#pragma unroll 1
  for (int it = 0; it < 16; ++it) {
    const int i = waveId * 16 + it;
    const int j = idx[(size_t)i * Kn + e];
    if (half == 0) {
#pragma unroll
      for (int f = 0; f < 64; ++f)
        feat[wv][e][f] = (_Float16)xin[(size_t)i * 64 + f];
    } else {
#pragma unroll
      for (int f = 0; f < 64; ++f)
        feat[wv][e][64 + f] =
            (_Float16)(xin[(size_t)j * 64 + f] - xin[(size_t)i * 64 + f]);
    }
    __syncthreads();
    v16h A[4];
#pragma unroll
    for (int ks = 0; ks < 4; ++ks) A[ks] = load_a(&feat[wv][0][0] + ks * 32, 128, lane);
#pragma unroll
    for (int t = 0; t < 4; ++t) {
      v8f acc = {};
#pragma unroll
      for (int ks = 0; ks < 4; ++ks) {
        const v16h bB = load_b(w1t + (size_t)t * 16 * 128 + ks * 32, 128, lane);
        acc = wmma_f16(A[ks], bB, acc);
      }
      const int col = t * 16 + e;
      const float bb = b1[col];
#pragma unroll
      for (int r = 0; r < 8; ++r)
        h1s[wv][r + 8 * half][col] = (_Float16)fmaxf(acc[r] + bb, 0.0f);
    }
    __syncthreads();
    const v16h H0 = load_a(&h1s[wv][0][0],      64, lane);
    const v16h H1 = load_a(&h1s[wv][0][0] + 32, 64, lane);
    float sqa = 0.0f;
#pragma unroll
    for (int t = 0; t < 4; ++t) {
      v8f acc = {};
      acc = wmma_f16(H0, wB2[t][0], acc);
      acc = wmma_f16(H1, wB2[t][1], acc);
      float mv = acc[0];
#pragma unroll
      for (int r = 1; r < 8; ++r) mv = fmaxf(mv, acc[r]);
      mv = fmaxf(mv, __shfl_xor(mv, 16));
      const int col = t * 16 + e;
      const float outv = mv + b2[col];
      if (half == 0) {
        embOut[(size_t)i * 64 + col]  = outv;
        embhOut[(size_t)i * 64 + col] = (_Float16)outv;
      }
      sqa += outv * outv;
    }
#pragma unroll
    for (int m = 8; m >= 1; m >>= 1) sqa += __shfl_xor(sqa, m);
    if (lane == 0) sqOut[i] = sqa;
    __syncthreads();
  }
}

// ---------------- Edge MLP: cat[x_src,x_dst] -> WMMA 128x128 relu -> dot w2 ---------
__global__ void edge_mlp_kernel(const _Float16* __restrict__ xh,
                                const int* __restrict__ idx,
                                const _Float16* __restrict__ w1t,
                                const float* __restrict__ b1,
                                const float* __restrict__ w2,
                                const float* __restrict__ b2,
                                float* __restrict__ out) {
  __shared__ __align__(16) _Float16 feat[8][16][128];
  const int lane = threadIdx.x & 31, wv = threadIdx.x >> 5;
  const int e = lane & 15, half = lane >> 4;
  const int waveId = blockIdx.x * 8 + wv;
  const float bias2 = b2[0];
  const size_t E = (size_t)BN * Kn;

#pragma unroll 1
  for (int it = 0; it < 16; ++it) {
    const int i = waveId * 16 + it;
    const int src = idx[(size_t)i * Kn + e];
    if (half == 0) {  // feat[e][0..63] = x[src]
      const v8h* sp = (const v8h*)(xh + (size_t)src * 64);
      v8h* dp = (v8h*)(&feat[wv][e][0]);
#pragma unroll
      for (int q = 0; q < 8; ++q) dp[q] = sp[q];
    } else {          // feat[e][64..127] = x[dst] (= node i)
      const v8h* sp = (const v8h*)(xh + (size_t)i * 64);
      v8h* dp = (v8h*)(&feat[wv][e][64]);
#pragma unroll
      for (int q = 0; q < 8; ++q) dp[q] = sp[q];
    }
    __syncthreads();
    v16h A[4];
#pragma unroll
    for (int ks = 0; ks < 4; ++ks) A[ks] = load_a(&feat[wv][0][0] + ks * 32, 128, lane);
    float part[8];
#pragma unroll
    for (int r = 0; r < 8; ++r) part[r] = 0.0f;
#pragma unroll
    for (int t = 0; t < 8; ++t) {
      v8f acc = {};
#pragma unroll
      for (int ks = 0; ks < 4; ++ks) {
        const v16h bB = load_b(w1t + (size_t)t * 16 * 128 + ks * 32, 128, lane);
        acc = wmma_f16(A[ks], bB, acc);
      }
      const int col = t * 16 + e;
      const float bb = b1[col], ww = w2[col];
#pragma unroll
      for (int r = 0; r < 8; ++r) part[r] += fmaxf(acc[r] + bb, 0.0f) * ww;
    }
#pragma unroll
    for (int r = 0; r < 8; ++r) {
#pragma unroll
      for (int m = 8; m >= 1; m >>= 1) part[r] += __shfl_xor(part[r], m);
    }
    if (e == 0) {  // lane 0 -> rows 0..7, lane 16 -> rows 8..15
#pragma unroll
      for (int r = 0; r < 8; ++r) {
        const int m = r + 8 * half;
        const size_t eg = (size_t)i * Kn + m;
        const float z = part[r] + bias2;
        out[eg]         = 1.0f / (1.0f + expf(-z));
        out[E + eg]     = (float)idx[(size_t)i * Kn + m];  // edge_index src
        out[2 * E + eg] = (float)i;                        // edge_index dst
      }
    }
    __syncthreads();
  }
}

// ---------------- host-side launch ----------------
extern "C" void kernel_launch(void* const* d_in, const int* in_sizes, int n_in,
                              void* d_out, int out_size, void* d_ws, size_t ws_size,
                              hipStream_t stream) {
  (void)in_sizes; (void)n_in; (void)out_size; (void)ws_size;

  const float* x      = (const float*)d_in[0];
  // d_in[1] = batch (unused; graphs are equal-sized)
  const float* c1_w1  = (const float*)d_in[2];
  const float* c1_b1  = (const float*)d_in[3];
  const float* c1_w2  = (const float*)d_in[4];
  const float* c1_b2  = (const float*)d_in[5];
  const float* c2_w1  = (const float*)d_in[6];
  const float* c2_b1  = (const float*)d_in[7];
  const float* c2_w2  = (const float*)d_in[8];
  const float* c2_b2  = (const float*)d_in[9];
  const float* mlp_w1 = (const float*)d_in[10];
  const float* mlp_b1 = (const float*)d_in[11];
  const float* mlp_w2 = (const float*)d_in[12];
  const float* mlp_b2 = (const float*)d_in[13];

  char* ws = (char*)d_ws;
  size_t off = 0;
  auto take = [&](size_t bytes) -> char* {
    off = (off + 255) & ~(size_t)255;
    char* p = ws + off;
    off += bytes;
    return p;
  };

  int*      idx1   = (int*)take((size_t)BN * Kn * sizeof(int));
  float*    emb1   = (float*)take((size_t)BN * 64 * sizeof(float));
  _Float16* embh1  = (_Float16*)take((size_t)BN * 64 * sizeof(_Float16));
  float*    sq1    = (float*)take((size_t)BN * sizeof(float));
  int*      idx2   = (int*)take((size_t)BN * Kn * sizeof(int));
  float*    emb2   = (float*)take((size_t)BN * 64 * sizeof(float));
  _Float16* embh2  = (_Float16*)take((size_t)BN * 64 * sizeof(_Float16));
  float*    sq2    = (float*)take((size_t)BN * sizeof(float));
  int*      idx3   = (int*)take((size_t)BN * Kn * sizeof(int));
  _Float16* c1w2t  = (_Float16*)take((size_t)64 * 64 * sizeof(_Float16));
  _Float16* c2w1t  = (_Float16*)take((size_t)64 * 128 * sizeof(_Float16));
  _Float16* c2w2t  = (_Float16*)take((size_t)64 * 64 * sizeof(_Float16));
  _Float16* mlpw1t = (_Float16*)take((size_t)128 * 128 * sizeof(_Float16));

  // 1) transposed f16 weights for WMMA B-operands
  transpose_f16_kernel<<<(64 * 64 + 255) / 256, 256, 0, stream>>>(c1_w2, c1w2t, 64, 64);
  transpose_f16_kernel<<<(128 * 64 + 255) / 256, 256, 0, stream>>>(c2_w1, c2w1t, 128, 64);
  transpose_f16_kernel<<<(64 * 64 + 255) / 256, 256, 0, stream>>>(c2_w2, c2w2t, 64, 64);
  transpose_f16_kernel<<<(128 * 128 + 255) / 256, 256, 0, stream>>>(mlp_w1, mlpw1t, 128, 128);

  // 2) kNN on raw 3-d features
  knn3_kernel<<<BN / 256, 256, 0, stream>>>(x, idx1);

  // 3) EdgeConv 1 (also emits f16 copy + squared norms for next kNN)
  conv1_kernel<<<BN / 128, 256, 0, stream>>>(x, idx1, c1_w1, c1_b1, c1w2t, c1_b2,
                                             emb1, embh1, sq1);

  // 4) kNN on 64-d embeddings (WMMA dot products, fused top-16; 4 waves/block,
  //    64 candidates per barrier round, rolled subtile loop -> no VGPR spills)
  knn64_kernel<<<BN / 128, 128, 0, stream>>>(embh1, sq1, idx2);

  // 5) EdgeConv 2
  conv2_kernel<<<BN / 128, 256, 0, stream>>>(emb1, idx2, c2w1t, c2_b1, c2w2t, c2_b2,
                                             emb2, embh2, sq2);

  // 6) final kNN graph
  knn64_kernel<<<BN / 128, 128, 0, stream>>>(embh2, sq2, idx3);

  // 7) edge MLP + sigmoid + edge_index emission
  edge_mlp_kernel<<<BN / 128, 256, 0, stream>>>(embh2, idx3, mlpw1t, mlp_b1,
                                                mlp_w2, mlp_b2, (float*)d_out);
}
